// MLP_20529943675402
// MI455X (gfx1250) — compile-verified, matrix operand order checked by reading, not run on hardware
//
#include <hip/hip_runtime.h>
#include <math.h>

typedef __attribute__((ext_vector_type(16))) _Float16 v16h;
typedef __attribute__((ext_vector_type(8)))  _Float16 v8h;
typedef __attribute__((ext_vector_type(8)))  float    v8f;

#define NROWS 12288
#define IS    512
#define HS    1024
#define OS    256
#define KEEP  31

// Load a 16x32 f16 WMMA A/B fragment from K-contiguous row-major storage.
// Layout (CDNA5 ISA, 16-bit A 16x32): lane L holds row (L&15);
// VGPR0-3 hold K = kq..kq+7 (2 halfs/VGPR), VGPR4-7 hold K = kq+16..kq+23,
// where kq = (L>>4)*8.  Works for global or LDS pointers (inlined).
__device__ __forceinline__ v16h frag_ld(const _Float16* base, int ld, int lane) {
    const int row = lane & 15;
    const int kq  = (lane >> 4) << 3;
    const _Float16* p = base + (size_t)row * ld + kq;
    v8h lo = *(const v8h*)(p);
    v8h hi = *(const v8h*)(p + 16);
    v16h r;
#pragma unroll
    for (int i = 0; i < 8; ++i) { r[i] = lo[i]; r[i + 8] = hi[i]; }
    return r;
}

__device__ __forceinline__ v8f wmma_f16(v16h a, v16h b, v8f c) {
    return __builtin_amdgcn_wmma_f32_16x16x32_f16(false, a, false, b,
                                                  (short)0, c, false, false);
}

// ---------------- Kernel 0: f32 -> f16 convert ----------------
__global__ __launch_bounds__(256) void cvt_f16_kernel(const float* __restrict__ s,
                                                      _Float16* __restrict__ d, int n) {
    int i = blockIdx.x * blockDim.x + threadIdx.x;
    int stride = gridDim.x * blockDim.x;
    for (; i < n; i += stride) d[i] = (_Float16)s[i];
}

// ---------------- Kernel 1: fused MLP + row-normalize ----------------
// One block = 256 threads = 8 waves handles 16 rows of the batch.
__global__ __launch_bounds__(256) void mlp_kernel(const _Float16* __restrict__ feat,
                                                  const _Float16* __restrict__ w1,
                                                  const _Float16* __restrict__ w2,
                                                  const float* __restrict__ b1,
                                                  const float* __restrict__ b2,
                                                  _Float16* __restrict__ emb) {
    __shared__ _Float16 hl[16][HS + 8];
    __shared__ float    nrm[16];

    const int tid  = threadIdx.x;
    const int wave = tid >> 5;
    const int lane = tid & 31;
    const int m0   = blockIdx.x * 16;
    const int ncol = lane & 15;
    const int mrow = (lane >> 4) << 3;

    // Stage 1: h = relu(F @ W1^T + b1); 64 column-tiles, 8 per wave,
    // processed as 4 pairs sharing the A fragment (dual WMMA chains).
    for (int p = 0; p < 4; ++p) {
        const int n0a = (wave + p * 16) * 16;
        const int n0b = (wave + 8 + p * 16) * 16;
        v8f c0 = {}, c1 = {};
        for (int k = 0; k < IS; k += 32) {
            v16h a  = frag_ld(feat + (size_t)m0 * IS + k, IS, lane);
            v16h b0 = frag_ld(w1 + (size_t)n0a * IS + k, IS, lane);
            v16h b1v = frag_ld(w1 + (size_t)n0b * IS + k, IS, lane);
            c0 = wmma_f16(a, b0, c0);
            c1 = wmma_f16(a, b1v, c1);
        }
        float biasa = b1[n0a + ncol];
        float biasb = b1[n0b + ncol];
#pragma unroll
        for (int v = 0; v < 8; ++v) {
            hl[mrow + v][n0a + ncol] = (_Float16)fmaxf(c0[v] + biasa, 0.0f);
            hl[mrow + v][n0b + ncol] = (_Float16)fmaxf(c1[v] + biasb, 0.0f);
        }
    }
    if (tid < 16) nrm[tid] = 0.0f;
    __syncthreads();

    // Stage 2: emb = h @ W2^T + b2 from LDS; 16 column-tiles, 2 per wave,
    // both run concurrently sharing the A fragment.
    const int n0a = wave * 16;
    const int n0b = (wave + 8) * 16;
    v8f c0 = {}, c1 = {};
    for (int k = 0; k < HS; k += 32) {
        v16h a  = frag_ld(&hl[0][0] + k, HS + 8, lane);
        v16h b0 = frag_ld(w2 + (size_t)n0a * HS + k, HS, lane);
        v16h b1v = frag_ld(w2 + (size_t)n0b * HS + k, HS, lane);
        c0 = wmma_f16(a, b0, c0);
        c1 = wmma_f16(a, b1v, c1);
    }
    {
        float biasa = b2[n0a + ncol];
        float biasb = b2[n0b + ncol];
#pragma unroll
        for (int v = 0; v < 8; ++v) {
            c0[v] += biasa;
            c1[v] += biasb;
            atomicAdd(&nrm[mrow + v], c0[v] * c0[v] + c1[v] * c1[v]);
        }
    }
    __syncthreads();

    // Stage 3: normalize rows, emit f16
#pragma unroll
    for (int v = 0; v < 8; ++v) {
        int m = mrow + v;
        float inv = 1.0f / fmaxf(sqrtf(nrm[m]), 1e-12f);
        emb[(size_t)(m0 + m) * OS + n0a + ncol] = (_Float16)(c0[v] * inv);
        emb[(size_t)(m0 + m) * OS + n0b + ncol] = (_Float16)(c1[v] * inv);
    }
}

// ---------------- Kernel 2: sim = E @ E^T (64x64 block tile) ----------------
// 8 waves, 16 output tiles: each wave owns 2 tiles in the same tile-row,
// sharing one A fragment -> two independent WMMA accumulate chains.
__global__ __launch_bounds__(256) void sim_kernel(const _Float16* __restrict__ emb,
                                                  float* __restrict__ out) {
    __shared__ _Float16 As[64][OS + 8];
    __shared__ _Float16 Bs[64][OS + 8];

    const int tid  = threadIdx.x;
    const int wave = tid >> 5;
    const int lane = tid & 31;
    const int mBase = blockIdx.x * 64;
    const int nBase = blockIdx.y * 64;

    // Stage 64 A-rows and 64 B-rows: each thread loads 64 halfs of each
    {
        int r = tid >> 2, o = (tid & 3) * 64;
        const v8h* gA = (const v8h*)(emb + (size_t)(mBase + r) * OS + o);
        const v8h* gB = (const v8h*)(emb + (size_t)(nBase + r) * OS + o);
#pragma unroll
        for (int j = 0; j < 8; ++j) {
            *(v8h*)&As[r][o + j * 8] = gA[j];
            *(v8h*)&Bs[r][o + j * 8] = gB[j];
        }
    }
    __syncthreads();

    const int tm  = wave >> 1;               // 0..3
    const int tn0 = (wave & 1) * 2;          // 0 or 2
    const int tn1 = tn0 + 1;
    v8f c0 = {}, c1 = {};
#pragma unroll
    for (int k = 0; k < OS; k += 32) {
        v16h a  = frag_ld(&As[tm * 16][0] + k, OS + 8, lane);
        v16h b0 = frag_ld(&Bs[tn0 * 16][0] + k, OS + 8, lane);
        v16h b1v = frag_ld(&Bs[tn1 * 16][0] + k, OS + 8, lane);
        c0 = wmma_f16(a, b0, c0);
        c1 = wmma_f16(a, b1v, c1);
    }
    const int m  = mBase + tm * 16 + ((lane >> 4) << 3);
    const int n0 = nBase + tn0 * 16 + (lane & 15);
    const int n1 = nBase + tn1 * 16 + (lane & 15);
#pragma unroll
    for (int v = 0; v < 8; ++v) {
        out[(size_t)(m + v) * NROWS + n0] = c0[v];
        out[(size_t)(m + v) * NROWS + n1] = c1[v];
    }
}

// ---------------- Kernel 3: per-row top-(K+1) threshold + elu, in place ----
__global__ __launch_bounds__(256) void topk_elu_kernel(float* __restrict__ out) {
    __shared__ __align__(16) float row[NROWS];   // 48 KB, fits (320 KB/WGP)
    __shared__ int   hist[256];
    __shared__ float cand[512];
    __shared__ int   ncand;
    __shared__ float s_lo, s_hi, thr;
    __shared__ int   s_R;

    const int tid = threadIdx.x;
    float* rp = out + (size_t)blockIdx.x * NROWS;

    // Load the full row into LDS (float4-coalesced)
    {
        const float4* g4 = (const float4*)rp;
        float4* s4 = (float4*)row;
        for (int i = tid; i < NROWS / 4; i += 256) s4[i] = g4[i];
    }
    if (tid == 0) { s_lo = -1.001f; s_hi = 1.001f; s_R = KEEP; ncand = 0; }
    __syncthreads();

    // Two 256-bucket refinement rounds: bucket width -> ~3e-5
    for (int pass = 0; pass < 2; ++pass) {
        float lo = s_lo, hi = s_hi;
        hist[tid] = 0;
        __syncthreads();
        float invw = 256.0f / (hi - lo);
        for (int i = tid; i < NROWS; i += 256) {
            float v = row[i];
            if (v >= lo && v < hi) {
                int b = (int)((v - lo) * invw);
                b = b > 255 ? 255 : b;
                atomicAdd(&hist[b], 1);
            }
        }
        __syncthreads();
        if (tid == 0) {
            int acc = 0, b = 255;
            for (; b > 0; --b) { acc += hist[b]; if (acc >= s_R) break; }
            if (acc < s_R) acc += hist[0];
            int above = acc - hist[b];       // strictly above chosen bucket
            s_R -= above;
            float w = (hi - lo) * (1.0f / 256.0f);
            s_lo = lo + b * w;
            s_hi = lo + (b + 1) * w;
        }
        __syncthreads();
    }

    // Collect candidates in final bucket, exact rank-select among them
    {
        float lo = s_lo, hi = s_hi;
        for (int i = tid; i < NROWS; i += 256) {
            float v = row[i];
            if (v >= lo && v < hi) {
                int idx = atomicAdd(&ncand, 1);
                if (idx < 512) cand[idx] = v;
            }
        }
    }
    if (tid == 0) thr = s_lo;                // fallback: keep whole bucket
    __syncthreads();
    {
        int nc = ncand < 512 ? ncand : 512;
        int r  = s_R;
        for (int j = tid; j < nc; j += 256) {
            float v = cand[j];
            int cg = 0, cge = 0;
            for (int i = 0; i < nc; ++i) {
                float u = cand[i];
                cg  += (u > v);
                cge += (u >= v);
            }
            if (cg < r && cge >= r) thr = v; // unique value; dup writes benign
        }
    }
    __syncthreads();

    // out = v>=thr ? elu(6v-6)+1 : elu(-6)+1  (6v-6<=0 => exp form)
    const float t = thr;
    const float dropped = 0.00247875217f;    // exp(-6)
    for (int i = tid; i < NROWS; i += 256) {
        float v = row[i];
        float res;
        if (v >= t) {
            float x = 6.0f * v - 6.0f;
            res = (x > 0.0f) ? (x + 1.0f) : expf(x);
        } else {
            res = dropped;
        }
        rp[i] = res;
    }
}

// ---------------- Host launcher ----------------
extern "C" void kernel_launch(void* const* d_in, const int* in_sizes, int n_in,
                              void* d_out, int out_size, void* d_ws, size_t ws_size,
                              hipStream_t stream) {
    const float* features = (const float*)d_in[0]; // [12288,512]
    const float* W1       = (const float*)d_in[1]; // [1024,512]
    const float* b1       = (const float*)d_in[2]; // [1024]
    const float* W2       = (const float*)d_in[3]; // [256,1024]
    const float* b2       = (const float*)d_in[4]; // [256]
    float* out = (float*)d_out;                    // [12288,12288]

    char* w = (char*)d_ws;
    _Float16* feat16 = (_Float16*)w;  w += (size_t)NROWS * IS * sizeof(_Float16);
    _Float16* w1_16  = (_Float16*)w;  w += (size_t)HS * IS * sizeof(_Float16);
    _Float16* w2_16  = (_Float16*)w;  w += (size_t)OS * HS * sizeof(_Float16);
    _Float16* emb16  = (_Float16*)w;  w += (size_t)NROWS * OS * sizeof(_Float16);

    cvt_f16_kernel<<<2048, 256, 0, stream>>>(features, feat16, NROWS * IS);
    cvt_f16_kernel<<<1024, 256, 0, stream>>>(W1, w1_16, HS * IS);
    cvt_f16_kernel<<<512, 256, 0, stream>>>(W2, w2_16, OS * HS);

    mlp_kernel<<<NROWS / 16, 256, 0, stream>>>(feat16, w1_16, w2_16, b1, b2, emb16);

    dim3 g(NROWS / 64, NROWS / 64);
    sim_kernel<<<g, 256, 0, stream>>>(emb16, out);

    topk_elu_kernel<<<NROWS, 256, 0, stream>>>(out);
}
